// MultiHeadAttention_59356448031218
// MI455X (gfx1250) — matrix operand order repaired and checked by reading, not compiled, not run on hardware
//
#include <hip/hip_runtime.h>
#include <hip/hip_bf16.h>

// ---------------------------------------------------------------------------
// Types for gfx1250 WMMA (wave32): v_wmma_f32_16x16x32_bf16
// ---------------------------------------------------------------------------
typedef __bf16 bf16;
typedef __attribute__((ext_vector_type(16))) __bf16 v16bf;
typedef __attribute__((ext_vector_type(8)))  __bf16 v8bf;
typedef __attribute__((ext_vector_type(8)))  float  v8f;
typedef int v4i_ __attribute__((vector_size(16)));  // b128 payload type

union Frag16 { v16bf v; v8bf h[2]; };

#define EPSF 1e-5f
#define D_   1024
#define NROW 4096   // B*S
#define HEADS 16
#define DH   64

// gfx1250 async global->LDS path (guarded: falls back to sync copy if the
// toolchain doesn't declare the builtins).
#if __has_builtin(__builtin_amdgcn_global_load_async_to_lds_b128) && \
    __has_builtin(__builtin_amdgcn_s_wait_asynccnt)
#define USE_ASYNC_LDS 1
#else
#define USE_ASYNC_LDS 0
#endif

__device__ __forceinline__ void stage16B(bf16* dst_lds, const bf16* src_glob) {
#if USE_ASYNC_LDS
  // emits global_load_async_to_lds_b128 (tracked by ASYNCcnt).
  // Signature (from hipcc diagnostic): (v4i AS1*, v4i AS3*, imm offset, imm cpol)
  __builtin_amdgcn_global_load_async_to_lds_b128(
      (__attribute__((address_space(1))) v4i_*)(unsigned long long)(uintptr_t)
          src_glob,
      (__attribute__((address_space(3))) v4i_*)(unsigned int)(uintptr_t)dst_lds,
      0, 0);
#else
  *(v8bf*)dst_lds = *(const v8bf*)src_glob;
#endif
}

__device__ __forceinline__ void async_fence() {
#if USE_ASYNC_LDS
  __builtin_amdgcn_s_wait_asynccnt(0);  // s_wait_asynccnt 0
#endif
}

__device__ __forceinline__ v8f wmma_bf16(v16bf a, v16bf b, v8f c) {
  // (neg_a, A, neg_b, B, c_mod, C, reuse_a, reuse_b)
  return __builtin_amdgcn_wmma_f32_16x16x32_bf16(false, a, false, b, (short)0, c,
                                                 false, false);
}

__device__ __forceinline__ float fast_rcp(float x) {
  return __builtin_amdgcn_rcpf(x);  // v_rcp_f32
}

// A fragment (16x32 bf16): lane<16: row=lane, K 0-7 & 16-23; lane>=16: K 8-15 & 24-31.
template <int PITCH>
__device__ __forceinline__ v16bf load_a_frag(const bf16* base, int lane) {
  int row = lane & 15;
  int kp  = (lane >> 4) * 8;
  Frag16 f;
  f.h[0] = *(const v8bf*)(base + row * PITCH + kp);
  f.h[1] = *(const v8bf*)(base + row * PITCH + kp + 16);
  return f.v;
}

// B fragment (32x16 bf16) loaded from a B-transposed [N][K] row-major LDS tile:
// lane: n = lane&15, K = (lane>>4)*8 .. +7 and 16+(lane>>4)*8 .. +7.
template <int PITCH>
__device__ __forceinline__ v16bf load_b_frag(const bf16* baseT, int lane) {
  int n  = lane & 15;
  int kb = (lane >> 4) * 8;
  Frag16 f;
  f.h[0] = *(const v8bf*)(baseT + n * PITCH + kb);
  f.h[1] = *(const v8bf*)(baseT + n * PITCH + kb + 16);
  return f.v;
}

// ---------------------------------------------------------------------------
// prep_x: X (f32) -> Xbf (bf16), xsq[row] = ||x_row||^2
// ---------------------------------------------------------------------------
__global__ __launch_bounds__(256) void prep_x_kernel(const float* __restrict__ X,
                                                     bf16* __restrict__ Xbf,
                                                     float* __restrict__ xsq) {
  int row = blockIdx.x;
  int t = threadIdx.x;
  float acc = 0.f;
#pragma unroll
  for (int i = 0; i < 4; ++i) {
    int idx = t + i * 256;
    float v = X[row * D_ + idx];
    Xbf[row * D_ + idx] = (bf16)v;
    acc += v * v;
  }
#pragma unroll
  for (int m = 1; m < 32; m <<= 1) acc += __shfl_xor(acc, m, 32);
  __shared__ float red[8];
  if ((t & 31) == 0) red[t >> 5] = acc;
  __syncthreads();
  if (t == 0) {
    float s = 0.f;
#pragma unroll
    for (int i = 0; i < 8; ++i) s += red[i];
    xsq[row] = s;
  }
}

// ---------------------------------------------------------------------------
// prep_w: W (f32, [K=D rows][N=D cols]) -> Wbf, colsq[j] += partial ||w_:,j||^2
// grid = (D/256 col groups, 8 row groups of 128); colsq pre-zeroed.
// ---------------------------------------------------------------------------
__global__ __launch_bounds__(256) void prep_w_kernel(const float* __restrict__ W,
                                                     bf16* __restrict__ Wbf,
                                                     float* __restrict__ colsq) {
  int col = blockIdx.x * 256 + threadIdx.x;
  int r0 = blockIdx.y * 128;
  float acc = 0.f;
  for (int r = r0; r < r0 + 128; ++r) {
    float v = W[r * D_ + col];
    Wbf[r * D_ + col] = (bf16)v;
    acc += v * v;
  }
  if (colsq) atomicAdd(&colsq[col], acc);
}

// ---------------------------------------------------------------------------
// GEMM: C = A(4096xD) * W(DxD). MODE 0: yat epilogue -> bf16 Y + head norms.
//       MODE 1: +bias -> f32 out.
// Block tile 128x64, 8 waves, wave tile 32x32, K-step 32.
// ---------------------------------------------------------------------------
template <int MODE>
__global__ __launch_bounds__(256) void yat_gemm_kernel(
    const bf16* __restrict__ A, const bf16* __restrict__ W,
    const float* __restrict__ rowsq, const float* __restrict__ colsq,
    const float* __restrict__ bias, const float* __restrict__ alpha,
    bf16* __restrict__ Ybf, float* __restrict__ Yf, float* __restrict__ yn) {
  __shared__ bf16 sA[128][40];   // [M][K] + pad
  __shared__ bf16 sBt[64][40];   // [N][K] + pad (W tile transposed)

  int tid = threadIdx.x, lane = tid & 31, wave = tid >> 5;
  int wm = (wave >> 1) * 32, wn = (wave & 1) * 32;
  int m0 = blockIdx.x * 128, n0 = blockIdx.y * 64;

  v8f acc[2][2] = {};

  for (int k0 = 0; k0 < D_; k0 += 32) {
    __syncthreads();
    {  // stage A tile: 128x32, 32B per thread (async -> LDS when available)
      int row = tid >> 1, ch = (tid & 1) * 16;
      const bf16* src = &A[(m0 + row) * D_ + k0 + ch];
      stage16B(&sA[row][ch], src);
      stage16B(&sA[row][ch + 8], src + 8);
      if (k0 + 32 < D_) __builtin_prefetch(src + 32, 0, 1);
    }
#pragma unroll
    for (int i = 0; i < 8; ++i) {  // stage W tile transposed: 32x64 -> [n][k]
      int idx = tid + 256 * i;
      int k = idx >> 6, n = idx & 63;
      sBt[n][k] = W[(k0 + k) * D_ + n0 + n];
    }
    async_fence();
    __syncthreads();

    v16bf af[2], bfm[2];
    af[0]  = load_a_frag<40>(&sA[wm][0], lane);
    af[1]  = load_a_frag<40>(&sA[wm + 16][0], lane);
    bfm[0] = load_b_frag<40>(&sBt[wn][0], lane);
    bfm[1] = load_b_frag<40>(&sBt[wn + 16][0], lane);
#pragma unroll
    for (int mi = 0; mi < 2; ++mi)
#pragma unroll
      for (int ni = 0; ni < 2; ++ni)
        acc[mi][ni] = wmma_bf16(af[mi], bfm[ni], acc[mi][ni]);
  }

  int lrow = (lane >> 4) * 8;
  int lcol = lane & 15;
  float scl = 1.0f;
  if constexpr (MODE == 0) {
    // (sqrt(dout)/log(1+dout))^alpha, dout = 1024
    scl = powf(32.0f / logf(1025.0f), alpha[0]);
  }

#pragma unroll
  for (int mi = 0; mi < 2; ++mi) {
#pragma unroll
    for (int r = 0; r < 8; ++r) {
      int row = m0 + wm + mi * 16 + lrow + r;
      float y2sum = 0.f;
#pragma unroll
      for (int ni = 0; ni < 2; ++ni) {
        int col = n0 + wn + ni * 16 + lcol;
        float d = acc[mi][ni][r];
        if constexpr (MODE == 0) {
          float dist = rowsq[row] + colsq[col] - 2.0f * d + EPSF;
          float y = (d * d) * fast_rcp(dist) + bias[col];
          y *= scl;
          Ybf[row * D_ + col] = (bf16)y;
          y2sum += y * y;
        } else {
          Yf[row * D_ + col] = d + bias[col];
        }
      }
      if constexpr (MODE == 0) {
        y2sum += __shfl_xor(y2sum, 1, 16);
        y2sum += __shfl_xor(y2sum, 2, 16);
        y2sum += __shfl_xor(y2sum, 4, 16);
        y2sum += __shfl_xor(y2sum, 8, 16);
        if (lcol == 0) atomicAdd(&yn[row * HEADS + (n0 >> 6)], y2sum);
      }
    }
  }
}

// ---------------------------------------------------------------------------
// Attention: per (b,h), flash-style online softmax of yat-transformed scores.
// Block = 8 waves x 16 query rows = 128 rows; grid = (S/128, B*H).
// ---------------------------------------------------------------------------
__global__ __launch_bounds__(256) void attn_kernel(
    const bf16* __restrict__ Q, const bf16* __restrict__ K,
    const bf16* __restrict__ V, const float* __restrict__ qn,
    const float* __restrict__ kn, bf16* __restrict__ O) {
  __shared__ bf16 sK[32][72];       // K chunk [key][feat] (== B^T layout for QK^T)
  __shared__ bf16 sVt[64][40];      // V chunk transposed [feat][key]
  __shared__ bf16 sP[8][16][40];    // per-wave P staging (C layout -> A frag)

  int tid = threadIdx.x, lane = tid & 31, wave = tid >> 5;
  int bh = blockIdx.y;
  int b = bh >> 4, h = bh & 15;
  int q0 = blockIdx.x * 128 + wave * 16;
  int lrow = (lane >> 4) * 8, lcol = lane & 15;

  // Q fragments live in registers for the whole kernel (16x64 = 2 A-frags)
  v16bf aq_[2];
  {
    long rowb = (long)(b * 1024 + q0 + lcol) * D_ + h * DH;
    int kp = (lane >> 4) * 8;
#pragma unroll
    for (int f = 0; f < 2; ++f) {
      Frag16 fr;
      fr.h[0] = *(const v8bf*)&Q[rowb + f * 32 + kp];
      fr.h[1] = *(const v8bf*)&Q[rowb + f * 32 + kp + 16];
      aq_[f] = fr.v;
    }
  }

  float ns[8], mx[8], l[8];
  v8f accO[4] = {};
#pragma unroll
  for (int r = 0; r < 8; ++r) {
    int row = b * 1024 + q0 + lrow + r;
    ns[r] = qn[row * HEADS + h] + kn[row * HEADS + h];  // query-row-only, per ref
    mx[r] = -3.0e38f;
    l[r] = 0.f;
  }

  for (int kc = 0; kc < 1024; kc += 32) {
    __syncthreads();
    {  // stage K chunk (async b128) and V chunk transposed (scalar)
      int kk = tid >> 3, fs = (tid & 7) * 8;
      stage16B(&sK[kk][fs], &K[(long)(b * 1024 + kc + kk) * D_ + h * DH + fs]);
      if (kc + 32 < 1024) {
        __builtin_prefetch(&K[(long)(b * 1024 + kc + 32 + kk) * D_ + h * DH + fs],
                           0, 1);
        __builtin_prefetch(&V[(long)(b * 1024 + kc + 32 + kk) * D_ + h * DH + fs],
                           0, 1);
      }
#pragma unroll
      for (int i = 0; i < 8; ++i) {
        int idx = tid + 256 * i;
        int vk = idx >> 6, f = idx & 63;
        sVt[f][vk] = V[(long)(b * 1024 + kc + vk) * D_ + h * DH + f];
      }
    }
    async_fence();
    __syncthreads();

    // scores S[16 q][32 keys] = Q(16x64) * K^T : hoist B frags, then 4 WMMAs
    v16bf bk_[2][2];
#pragma unroll
    for (int ks = 0; ks < 2; ++ks)
#pragma unroll
      for (int fs = 0; fs < 2; ++fs)
        bk_[ks][fs] = load_b_frag<72>(&sK[ks * 16][fs * 32], lane);
    v8f cs[2] = {};
#pragma unroll
    for (int ks = 0; ks < 2; ++ks)
#pragma unroll
      for (int fs = 0; fs < 2; ++fs)
        cs[ks] = wmma_bf16(aq_[fs], bk_[ks][fs], cs[ks]);

    // yat transform + online softmax per row (fully unrolled: no v_movrels)
#pragma unroll
    for (int r = 0; r < 8; ++r) {
      float s0 = cs[0][r], s1 = cs[1][r];
      float q0_ = s0 * s0, q1_ = s1 * s1;
      float w0 = q0_ * fast_rcp(ns[r] - 2.0f * q0_ + EPSF);
      float w1 = q1_ * fast_rcp(ns[r] - 2.0f * q1_ + EPSF);
      float cm = fmaxf(w0, w1);
#pragma unroll
      for (int s = 1; s < 16; s <<= 1) cm = fmaxf(cm, __shfl_xor(cm, s, 16));
      float nm = fmaxf(mx[r], cm);
      float corr = __expf(mx[r] - nm);
      float p0 = __expf(w0 - nm), p1 = __expf(w1 - nm);
      float ps = p0 + p1;
#pragma unroll
      for (int s = 1; s < 16; s <<= 1) ps += __shfl_xor(ps, s, 16);
      l[r] = l[r] * corr + ps;
      mx[r] = nm;
#pragma unroll
      for (int g = 0; g < 4; ++g) accO[g][r] *= corr;
      sP[wave][lrow + r][lcol] = (bf16)p0;       // C-layout -> LDS
      sP[wave][lrow + r][lcol + 16] = (bf16)p1;
    }

    // O += P(16x32) * V(32x64): hoist A + 4 B frags, then 4 WMMAs back-to-back
    v16bf pf = load_a_frag<40>(&sP[wave][0][0], lane);
    v16bf bv_[4];
#pragma unroll
    for (int g = 0; g < 4; ++g)
      bv_[g] = load_b_frag<40>(&sVt[g * 16][0], lane);
#pragma unroll
    for (int g = 0; g < 4; ++g) accO[g] = wmma_bf16(pf, bv_[g], accO[g]);
  }

#pragma unroll
  for (int g = 0; g < 4; ++g)
#pragma unroll
    for (int r = 0; r < 8; ++r) {
      float o = accO[g][r] * fast_rcp(l[r]);
      long row = b * 1024 + q0 + lrow + r;
      int col = h * DH + g * 16 + lcol;
      O[row * D_ + col] = (bf16)o;
    }
}

// ---------------------------------------------------------------------------
// Launch
// ---------------------------------------------------------------------------
extern "C" void kernel_launch(void* const* d_in, const int* in_sizes, int n_in,
                              void* d_out, int out_size, void* d_ws,
                              size_t ws_size, hipStream_t stream) {
  const float* X  = (const float*)d_in[0];
  const float* wq = (const float*)d_in[1];
  const float* bq = (const float*)d_in[2];
  const float* aq = (const float*)d_in[3];
  const float* wk = (const float*)d_in[4];
  const float* bk = (const float*)d_in[5];
  const float* ak = (const float*)d_in[6];
  const float* wv = (const float*)d_in[7];
  const float* bv = (const float*)d_in[8];
  const float* av = (const float*)d_in[9];
  const float* wo = (const float*)d_in[10];
  const float* bo = (const float*)d_in[11];

  char* p = (char*)d_ws;
  auto carve = [&](size_t bytes) -> void* {
    void* r = (void*)p;
    p += (bytes + 255) & ~(size_t)255;
    return r;
  };
  bf16* Xbf = (bf16*)carve((size_t)NROW * D_ * 2);
  bf16* Wqb = (bf16*)carve((size_t)D_ * D_ * 2);
  bf16* Wkb = (bf16*)carve((size_t)D_ * D_ * 2);
  bf16* Wvb = (bf16*)carve((size_t)D_ * D_ * 2);
  bf16* Wob = (bf16*)carve((size_t)D_ * D_ * 2);
  float* xsq = (float*)carve((size_t)NROW * 4);
  float* csq_q = (float*)carve((size_t)D_ * 4);  // csq_q/k/v contiguous (4KB each)
  float* csq_k = (float*)carve((size_t)D_ * 4);
  float* csq_v = (float*)carve((size_t)D_ * 4);
  bf16* Qb = (bf16*)carve((size_t)NROW * D_ * 2);
  bf16* Kb = (bf16*)carve((size_t)NROW * D_ * 2);
  bf16* Vb = (bf16*)carve((size_t)NROW * D_ * 2);
  bf16* Ob = (bf16*)carve((size_t)NROW * D_ * 2);
  float* qn = (float*)carve((size_t)NROW * HEADS * 4 * 3);  // qn,kn,vn contiguous
  float* kn = qn + (size_t)NROW * HEADS;
  float* vn = kn + (size_t)NROW * HEADS;

  (void)hipMemsetAsync(qn, 0, (size_t)NROW * HEADS * 4 * 3, stream);
  (void)hipMemsetAsync(csq_q, 0, (size_t)D_ * 4 * 3, stream);

  prep_x_kernel<<<NROW, 256, 0, stream>>>(X, Xbf, xsq);
  dim3 gw(D_ / 256, 8);
  prep_w_kernel<<<gw, 256, 0, stream>>>(wq, Wqb, csq_q);
  prep_w_kernel<<<gw, 256, 0, stream>>>(wk, Wkb, csq_k);
  prep_w_kernel<<<gw, 256, 0, stream>>>(wv, Wvb, csq_v);
  prep_w_kernel<<<gw, 256, 0, stream>>>(wo, Wob, nullptr);

  dim3 gg(NROW / 128, D_ / 64);
  yat_gemm_kernel<0><<<gg, 256, 0, stream>>>(Xbf, Wqb, xsq, csq_q, bq, aq, Qb,
                                             nullptr, qn);
  yat_gemm_kernel<0><<<gg, 256, 0, stream>>>(Xbf, Wkb, xsq, csq_k, bk, ak, Kb,
                                             nullptr, kn);
  yat_gemm_kernel<0><<<gg, 256, 0, stream>>>(Xbf, Wvb, xsq, csq_v, bv, av, Vb,
                                             nullptr, vn);

  attn_kernel<<<dim3(1024 / 128, 4 * HEADS), 256, 0, stream>>>(Qb, Kb, Vb, qn,
                                                               kn, Ob);

  yat_gemm_kernel<1><<<gg, 256, 0, stream>>>(Ob, Wob, nullptr, nullptr, bo,
                                             nullptr, nullptr, (float*)d_out,
                                             nullptr);
  (void)in_sizes; (void)n_in; (void)out_size; (void)ws_size;
}
